// SkillDetector_27917287424339
// MI455X (gfx1250) — matrix operand order repaired and checked by reading, not compiled, not run on hardware
//
#include <hip/hip_runtime.h>
#include <hip/hip_bf16.h>
#include <math.h>

// ---------------------------------------------------------------------------
// SkillDetector for MI455X (gfx1250, wave32, WMMA)
//   N=8192, D=512, H=256, MAX_ITERS=10
// ---------------------------------------------------------------------------

#define NN 8192
#define DD 512
#define HH 256
#define ITERS 10

typedef __attribute__((ext_vector_type(16))) _Float16 v16h;
typedef __attribute__((ext_vector_type(8)))  _Float16 v8h;
typedef __attribute__((ext_vector_type(8)))  float    v8f;

#define WMMA_F16(a, b, c) \
  __builtin_amdgcn_wmma_f32_16x16x32_f16(false, (a), false, (b), (short)0, (c), false, false)

// ---------------- WMMA fragment loaders (ISA 7.12.2 layouts) ----------------

// A fragment: 16x32 f16 tile at (m0,k0) of row-major [M x lda]
// lanes 0-15: M=lane,  K = k0 + {0..7, 16..23}
// lanes 16-31: M=lane-16, K = k0 + {8..15, 24..31}
__device__ __forceinline__ v16h frag_a(const _Float16* A, int lda, int m0, int k0) {
  int lane = threadIdx.x & 31;
  int m  = m0 + (lane & 15);
  int kb = (lane & 16) ? 8 : 0;
  const _Float16* p = A + (size_t)m * lda + (k0 + kb);
  __builtin_prefetch((const void*)(p + 32), 0, 3);   // next K-tile -> global_prefetch_b8
  v8h lo = *(const v8h*)p;          // K = kb+0 .. kb+7
  v8h hi = *(const v8h*)(p + 16);   // K = kb+16 .. kb+23
  v16h f;
#pragma unroll
  for (int i = 0; i < 8; ++i) { f[i] = lo[i]; f[i + 8] = hi[i]; }
  return f;
}

// B fragment where B[k][n] = X[n][k], X row-major [N x ldx]
// (used for transposed weights and for the A*A^T sim GEMM).
// lanes 0-15: K = k0 + {0..7, 16..23} ; lanes 16-31: K = k0 + {8..15, 24..31} ; N = n0+lane%16.
// Per lane this is two runs of 8 contiguous f16 along k -> two b128 loads.
__device__ __forceinline__ v16h frag_bT(const _Float16* X, int ldx, int k0, int n0) {
  int lane = threadIdx.x & 31;
  int n  = n0 + (lane & 15);
  int kb = (lane & 16) ? 8 : 0;
  const _Float16* p = X + (size_t)n * ldx + (k0 + kb);
  __builtin_prefetch((const void*)(p + 32), 0, 3);   // next K-tile
  v8h lo = *(const v8h*)p;          // K = kb+0 .. kb+7
  v8h hi = *(const v8h*)(p + 16);   // K = kb+16 .. kb+23
  v16h f;
#pragma unroll
  for (int i = 0; i < 8; ++i) { f[i] = lo[i]; f[i + 8] = hi[i]; }
  return f;
}

// -------------------------- C-tile store helpers ----------------------------
// C VGPR r: lanes 0-15 -> row m0+r, lanes 16-31 -> row m0+r+8, col n0+lane%16

__device__ __forceinline__ void store_gelu_f16(const v8f& c, _Float16* C, int ldc,
                                               const float* bias, int m0, int n0) {
  int lane = threadIdx.x & 31;
  int n  = n0 + (lane & 15);
  int mo = (lane & 16) ? 8 : 0;
  float bv = bias[n];
#pragma unroll
  for (int r = 0; r < 8; ++r) {
    float x = c[r] + bv;
    float g = 0.5f * x * (1.0f + erff(x * 0.70710678118654752f));  // exact GELU
    C[(size_t)(m0 + mo + r) * ldc + n] = (_Float16)g;
  }
}

__device__ __forceinline__ void store_bias_f32(const v8f& c, float* C, int ldc,
                                               const float* bias, int m0, int n0) {
  int lane = threadIdx.x & 31;
  int n  = n0 + (lane & 15);
  int mo = (lane & 16) ? 8 : 0;
  float bv = bias[n];
#pragma unroll
  for (int r = 0; r < 8; ++r)
    C[(size_t)(m0 + mo + r) * ldc + n] = c[r] + bv;
}

__device__ __forceinline__ void store_adj_bits(const v8f& c, float thr,
                                               unsigned short* p16, int u16PerRow,
                                               int m0, int tj) {
  int lane = threadIdx.x & 31;
#pragma unroll
  for (int r = 0; r < 8; ++r) {
    unsigned m = __builtin_amdgcn_ballot_w32(c[r] > thr);
    if (lane == 0) {
      p16[(size_t)(m0 + r)     * u16PerRow + tj] = (unsigned short)(m & 0xFFFFu);
      p16[(size_t)(m0 + r + 8) * u16PerRow + tj] = (unsigned short)(m >> 16);
    }
  }
}

// ---------------------------- elementwise / prep ----------------------------

__global__ void k_cvt(const float* __restrict__ src, _Float16* __restrict__ dst, int n) {
  int i = blockIdx.x * blockDim.x + threadIdx.x;
  if (i < n) dst[i] = (_Float16)src[i];
}

// transpose + convert: src [K x N] f32 row-major -> dst [N x K] f16 row-major
__global__ void k_cvt_t(const float* __restrict__ src, _Float16* __restrict__ dst,
                        int K, int N) {
  int i = blockIdx.x * blockDim.x + threadIdx.x;   // output-linear: writes coalesced
  if (i < K * N) {
    int n = i / K, k = i - n * K;
    dst[i] = (_Float16)src[(size_t)k * N + n];
  }
}

// ------------------------------- encoder GEMMs ------------------------------
// One 32x32 output tile per wave (2x2 register blocking), 8 waves / block.

// h = gelu(patterns @ W1 + b1), output f16 [M x N]; Bt = W1^T [N x K] f16
__global__ void k_gemm_gelu(const _Float16* __restrict__ A, const _Float16* __restrict__ Bt,
                            const float* __restrict__ bias, _Float16* __restrict__ C,
                            int M, int N, int K) {
  int wave   = blockIdx.x * (blockDim.x >> 5) + (threadIdx.x >> 5);
  int ntiles = N >> 5;
  int tm = wave / ntiles, tn = wave % ntiles;
  int m0 = tm * 32, n0 = tn * 32;
  if (m0 >= M) return;

  v8f c00 = {}, c01 = {}, c10 = {}, c11 = {};
  for (int k0 = 0; k0 < K; k0 += 32) {
    v16h a0 = frag_a(A, K, m0,      k0);
    v16h a1 = frag_a(A, K, m0 + 16, k0);
    v16h b0 = frag_bT(Bt, K, k0, n0);
    v16h b1 = frag_bT(Bt, K, k0, n0 + 16);
    c00 = WMMA_F16(a0, b0, c00);
    c01 = WMMA_F16(a0, b1, c01);
    c10 = WMMA_F16(a1, b0, c10);
    c11 = WMMA_F16(a1, b1, c11);
  }
  store_gelu_f16(c00, C, N, bias, m0,      n0);
  store_gelu_f16(c01, C, N, bias, m0,      n0 + 16);
  store_gelu_f16(c10, C, N, bias, m0 + 16, n0);
  store_gelu_f16(c11, C, N, bias, m0 + 16, n0 + 16);
}

// encoded = h @ W2 + b2, output f32 [M x N]; Bt = W2^T [N x K] f16
__global__ void k_gemm_bias(const _Float16* __restrict__ A, const _Float16* __restrict__ Bt,
                            const float* __restrict__ bias, float* __restrict__ C,
                            int M, int N, int K) {
  int wave   = blockIdx.x * (blockDim.x >> 5) + (threadIdx.x >> 5);
  int ntiles = N >> 5;
  int tm = wave / ntiles, tn = wave % ntiles;
  int m0 = tm * 32, n0 = tn * 32;
  if (m0 >= M) return;

  v8f c00 = {}, c01 = {}, c10 = {}, c11 = {};
  for (int k0 = 0; k0 < K; k0 += 32) {
    v16h a0 = frag_a(A, K, m0,      k0);
    v16h a1 = frag_a(A, K, m0 + 16, k0);
    v16h b0 = frag_bT(Bt, K, k0, n0);
    v16h b1 = frag_bT(Bt, K, k0, n0 + 16);
    c00 = WMMA_F16(a0, b0, c00);
    c01 = WMMA_F16(a0, b1, c01);
    c10 = WMMA_F16(a1, b0, c10);
    c11 = WMMA_F16(a1, b1, c11);
  }
  store_bias_f32(c00, C, N, bias, m0,      n0);
  store_bias_f32(c01, C, N, bias, m0,      n0 + 16);
  store_bias_f32(c10, C, N, bias, m0 + 16, n0);
  store_bias_f32(c11, C, N, bias, m0 + 16, n0 + 16);
}

// ------------------------------- normalization ------------------------------

__global__ void k_normalize(const float* __restrict__ E, _Float16* __restrict__ Xn,
                            int N, int D) {
  int row = blockIdx.x;
  int t = threadIdx.x;                       // 256 threads
  const float* e = E + (size_t)row * D;
  float s = 0.f;
  for (int d = t; d < D; d += blockDim.x) { float v = e[d]; s += v * v; }
  __shared__ float red[256];
  red[t] = s; __syncthreads();
  for (int st = 128; st; st >>= 1) { if (t < st) red[t] += red[t + st]; __syncthreads(); }
  float inv = 1.0f / fmaxf(sqrtf(red[0]), 1e-8f);
  for (int d = t; d < D; d += blockDim.x)
    Xn[(size_t)row * D + d] = (_Float16)(e[d] * inv);
}

// -------------------- sim = Xn @ Xn^T  ->  adjacency bitmap -----------------
// adj: NN rows of NN bits (NN/32 u32 words per row).
// One wave per 32x32 tile of sim; 4 WMMAs per K-step; operands L2-resident.

__global__ void k_sim_adj(const _Float16* __restrict__ X, unsigned* __restrict__ adj,
                          const float* __restrict__ sim_thr_p, int N, int K) {
  float thr = *sim_thr_p;
  int wave   = blockIdx.x * (blockDim.x >> 5) + (threadIdx.x >> 5);
  int ntiles = N >> 5;
  int ti = wave / ntiles, tj = wave % ntiles;
  if (ti >= ntiles) return;
  int m0 = ti * 32, n0 = tj * 32;

  v8f c00 = {}, c01 = {}, c10 = {}, c11 = {};
  for (int k0 = 0; k0 < K; k0 += 32) {
    v16h a0 = frag_a(X, K, m0,      k0);
    v16h a1 = frag_a(X, K, m0 + 16, k0);
    v16h b0 = frag_bT(X, K, k0, n0);
    v16h b1 = frag_bT(X, K, k0, n0 + 16);
    c00 = WMMA_F16(a0, b0, c00);
    c01 = WMMA_F16(a0, b1, c01);
    c10 = WMMA_F16(a1, b0, c10);
    c11 = WMMA_F16(a1, b1, c11);
  }

  unsigned short* p16 = (unsigned short*)adj;
  int u16PerRow = N >> 4;
  int tj0 = n0 >> 4;
  store_adj_bits(c00, thr, p16, u16PerRow, m0,      tj0);
  store_adj_bits(c01, thr, p16, u16PerRow, m0,      tj0 + 1);
  store_adj_bits(c10, thr, p16, u16PerRow, m0 + 16, tj0);
  store_adj_bits(c11, thr, p16, u16PerRow, m0 + 16, tj0 + 1);
}

// ------------------------------- peel iteration -----------------------------

__global__ void k_init_used(unsigned* __restrict__ used) { used[threadIdx.x] = 0u; }

// nbr[i] = used(i) ? 0 : popcount(adj[i] & ~used)
__global__ void k_nbr(const unsigned* __restrict__ adj, const unsigned* __restrict__ used,
                      int* __restrict__ nbr, int N) {
  int row = blockIdx.x;
  int t = threadIdx.x;                       // 256 threads == N/32 words
  int c = __popc(adj[(size_t)row * (N >> 5) + t] & ~used[t]);
  __shared__ int red[256];
  red[t] = c; __syncthreads();
  for (int st = 128; st; st >>= 1) { if (t < st) red[t] += red[t + st]; __syncthreads(); }
  if (t == 0) {
    bool usedRow = (used[row >> 5] >> (row & 31)) & 1u;
    nbr[row] = usedRow ? 0 : red[0];
  }
}

// first-max argmax + active flag
__global__ void k_argmax(const int* __restrict__ nbr, const float* __restrict__ det_thr_p,
                         int* __restrict__ center_p, int* __restrict__ active_p, int N) {
  int t = threadIdx.x;
  int bestV = -1, bestI = 0;
  for (int i = t; i < N; i += blockDim.x) {
    int v = nbr[i];
    if (v > bestV) { bestV = v; bestI = i; }
  }
  __shared__ int rv[256], ri[256];
  rv[t] = bestV; ri[t] = bestI; __syncthreads();
  for (int st = 128; st; st >>= 1) {
    if (t < st) {
      if (rv[t + st] > rv[t] || (rv[t + st] == rv[t] && ri[t + st] < ri[t])) {
        rv[t] = rv[t + st]; ri[t] = ri[t + st];
      }
    }
    __syncthreads();
  }
  if (t == 0) {
    int minc = (int)(*det_thr_p * 3.0f);
    if (minc < 3) minc = 3;
    *center_p = ri[0];
    *active_p = (rv[0] >= minc) ? 1 : 0;
  }
}

// S = sum of clamped rewards over cluster; write flags[iter]
__global__ void k_prep(const unsigned* __restrict__ adj, const unsigned* __restrict__ used,
                       const float* __restrict__ rewards, const int* __restrict__ center_p,
                       const int* __restrict__ active_p, float* __restrict__ Ssum,
                       float* __restrict__ out, int iter, int N, int D) {
  int t = threadIdx.x;
  int center = *center_p;
  int active = *active_p;
  const unsigned* arow = adj + (size_t)center * (N >> 5);
  float s = 0.f;
  if (active) {
    for (int j = t; j < N; j += blockDim.x) {
      unsigned ab = (arow[j >> 5] >> (j & 31)) & 1u;
      unsigned ub = (used[j >> 5] >> (j & 31)) & 1u;
      if (ab && !ub) s += fmaxf(rewards[j], 1e-8f);
    }
  }
  __shared__ float red[256];
  red[t] = s; __syncthreads();
  for (int st = 128; st; st >>= 1) { if (t < st) red[t] += red[t + st]; __syncthreads(); }
  if (t == 0) {
    *Ssum = red[0];
    out[(size_t)ITERS * D + iter] = active ? 1.0f : 0.0f;   // flags
  }
}

// per-block partial centroid over 256 rows; then update used bits (block-local)
__global__ void k_accum(const unsigned* __restrict__ adj, unsigned* __restrict__ used,
                        const float* __restrict__ rewards, const float* __restrict__ E,
                        const int* __restrict__ center_p, const int* __restrict__ active_p,
                        const float* __restrict__ Ssum, float* __restrict__ part,
                        int N, int D) {
  int t = threadIdx.x;                       // 256 threads: d = t and t+256
  int j0 = blockIdx.x * 256;
  int center = *center_p;
  int active = *active_p;
  float S = fmaxf(*Ssum, 1e-8f);
  const unsigned* arow = adj + (size_t)center * (N >> 5);

  float acc0 = 0.f, acc1 = 0.f;
  if (active) {
    for (int jj = 0; jj < 256; ++jj) {
      int j = j0 + jj;
      unsigned ab = (arow[j >> 5] >> (j & 31)) & 1u;
      unsigned ub = (used[j >> 5] >> (j & 31)) & 1u;
      if (ab && !ub) {
        float w = fmaxf(rewards[j], 1e-8f) / S;
        const float* e = E + (size_t)j * D;
        acc0 += w * e[t];
        acc1 += w * e[t + 256];
      }
    }
  }
  part[(size_t)blockIdx.x * D + t]       = acc0;
  part[(size_t)blockIdx.x * D + t + 256] = acc1;

  __syncthreads();  // all used-reads above complete before we mutate used

  int j = j0 + t;
  unsigned ab = (arow[j >> 5] >> (j & 31)) & 1u;
  unsigned ub = (used[j >> 5] >> (j & 31)) & 1u;
  unsigned bit = (active && ab && !ub) ? 1u : 0u;
  unsigned mask = __builtin_amdgcn_ballot_w32((int)bit);
  if ((t & 31) == 0) {
    int word = (j0 >> 5) + (t >> 5);       // block-exclusive word
    used[word] |= mask;
  }
}

// deterministic fixed-order reduction of 32 partials -> centroid row
__global__ void k_cfinal(const float* __restrict__ part, float* __restrict__ out,
                         int iter, int D) {
  int d = blockIdx.x * blockDim.x + threadIdx.x;
  if (d < D) {
    float s = 0.f;
#pragma unroll
    for (int b = 0; b < 32; ++b) s += part[(size_t)b * D + d];
    out[(size_t)iter * D + d] = s;
  }
}

// --------------------------------- launcher ---------------------------------

extern "C" void kernel_launch(void* const* d_in, const int* in_sizes, int n_in,
                              void* d_out, int out_size, void* d_ws, size_t ws_size,
                              hipStream_t stream) {
  (void)in_sizes; (void)n_in; (void)out_size; (void)ws_size;

  const float* patterns = (const float*)d_in[0];   // [NN, DD]
  const float* rewards  = (const float*)d_in[1];   // [NN]
  const float* W1       = (const float*)d_in[2];   // [DD, HH]
  const float* b1       = (const float*)d_in[3];   // [HH]
  const float* W2       = (const float*)d_in[4];   // [HH, DD]
  const float* b2       = (const float*)d_in[5];   // [DD]
  const float* sim_thr  = (const float*)d_in[6];   // scalar
  const float* det_thr  = (const float*)d_in[7];   // scalar
  float* out = (float*)d_out;                      // [ITERS*DD] centroids + [ITERS] flags

  char* ws = (char*)d_ws;
  size_t off = 0;
  _Float16* pat_h = (_Float16*)(ws + off); off += (size_t)NN * DD * 2;         // 8 MB
  _Float16* W1t   = (_Float16*)(ws + off); off += (size_t)DD * HH * 2;         // 256 KB (W1^T [HH x DD])
  _Float16* W2t   = (_Float16*)(ws + off); off += (size_t)HH * DD * 2;         // 256 KB (W2^T [DD x HH])
  _Float16* h_h   = (_Float16*)(ws + off); off += (size_t)NN * HH * 2;         // 4 MB
  float*    enc   = (float*)   (ws + off); off += (size_t)NN * DD * 4;         // 16 MB
  _Float16* encn  = (_Float16*)(ws + off); off += (size_t)NN * DD * 2;         // 8 MB
  unsigned* adj   = (unsigned*)(ws + off); off += (size_t)NN * (NN / 32) * 4;  // 8 MB
  int*      nbr   = (int*)     (ws + off); off += (size_t)NN * 4;              // 32 KB
  unsigned* used  = (unsigned*)(ws + off); off += (size_t)(NN / 32) * 4;       // 1 KB
  float*    part  = (float*)   (ws + off); off += (size_t)32 * DD * 4;         // 64 KB
  float*    Ssum  = (float*)   (ws + off); off += 256;
  int*      centr = (int*)     (ws + off); off += 256;
  int*      activ = (int*)     (ws + off); off += 256;

  // 1) f32 -> f16 conversions; weights transposed so all B fragments are
  //    contiguous-per-lane b128 loads
  k_cvt  <<<(NN * DD + 255) / 256, 256, 0, stream>>>(patterns, pat_h, NN * DD);
  k_cvt_t<<<(DD * HH + 255) / 256, 256, 0, stream>>>(W1, W1t, DD, HH);
  k_cvt_t<<<(HH * DD + 255) / 256, 256, 0, stream>>>(W2, W2t, HH, DD);

  // 2) encoder: h = gelu(patterns @ W1 + b1) ; encoded = h @ W2 + b2
  //    one 32x32 tile per wave (2x2 WMMA blocking), 8 waves / block
  k_gemm_gelu<<<(NN / 32) * (HH / 32) / 8, 256, 0, stream>>>(pat_h, W1t, b1, h_h, NN, HH, DD);
  k_gemm_bias<<<(NN / 32) * (DD / 32) / 8, 256, 0, stream>>>(h_h, W2t, b2, enc, NN, DD, HH);

  // 3) row-normalize -> f16
  k_normalize<<<NN, 256, 0, stream>>>(enc, encn, NN, DD);

  // 4) sim > thr adjacency bitmap via WMMA A*A^T (8 MB, L2-resident)
  k_sim_adj<<<(NN / 32) * (NN / 32) / 8, 256, 0, stream>>>(encn, adj, sim_thr, NN, DD);

  // 5) greedy peel, 10 iterations on the bitmap
  k_init_used<<<1, 256, 0, stream>>>(used);
  for (int it = 0; it < ITERS; ++it) {
    k_nbr   <<<NN, 256, 0, stream>>>(adj, used, nbr, NN);
    k_argmax<<<1, 256, 0, stream>>>(nbr, det_thr, centr, activ, NN);
    k_prep  <<<1, 256, 0, stream>>>(adj, used, rewards, centr, activ, Ssum, out, it, NN, DD);
    k_accum <<<NN / 256, 256, 0, stream>>>(adj, used, rewards, enc, centr, activ, Ssum, part, NN, DD);
    k_cfinal<<<(DD + 255) / 256, 256, 0, stream>>>(part, out, it, DD);
  }
}
